// RTDETRTransformer_66271345377386
// MI455X (gfx1250) — compile-verified, matrix operand order checked by reading, not run on hardware
//
#include <hip/hip_runtime.h>

// ---------------- problem constants ----------------
#define Bb    8
#define NTOK  8700
#define Cc    256
#define Qq    300
#define Hh    8
#define Dd    32
#define HIDN  1024
#define NPAD  8704   /* 32*272 : padded key count   */
#define QPAD  304    /* 16*19  : padded query count */
#define QTILES 19
#define BN    (Bb*NTOK)   /* 69600 */
#define BQ    (Bb*Qq)     /* 2400  */
#define ALDS_STRIDE 40    /* padded LDS stride (elements) -> conflict-free frag reads */

typedef __attribute__((ext_vector_type(16))) __bf16 v16bf;
typedef __attribute__((ext_vector_type(8)))  float  v8f;
typedef __attribute__((ext_vector_type(4)))  unsigned int u32x4;
typedef __attribute__((ext_vector_type(8)))  unsigned int u32x8;

__device__ __forceinline__ unsigned short f2bf(float f) {
  unsigned int u = __float_as_uint(f);
  u += 0x7fffu + ((u >> 16) & 1u);          // round-to-nearest-even
  return (unsigned short)(u >> 16);
}

// Load a 16x32 bf16 A/B WMMA fragment from row-major storage (row stride `ld`
// elements). CDNA5 ISA 7.12.2 layout: lane (l&15) holds row M=l&15;
// lanes 0-15 carry K = 0..7 and 16..23, lanes 16-31 carry K = 8..15 and 24..31.
// -> per lane: two contiguous 16-byte loads at element offsets (hf*8) and (16+hf*8).
__device__ __forceinline__ v16bf load_frag(const unsigned short* base, int ld, int lane) {
  const int r  = lane & 15;
  const int hf = (lane >> 4) & 1;
  const unsigned short* p = base + (size_t)r * ld + hf * 8;
  u32x4 lo = *(const u32x4*)(p);
  u32x4 hi = *(const u32x4*)(p + 16);
  u32x8 t;
  t[0] = lo[0]; t[1] = lo[1]; t[2] = lo[2]; t[3] = lo[3];
  t[4] = hi[0]; t[5] = hi[1]; t[6] = hi[2]; t[7] = hi[3];
  return __builtin_bit_cast(v16bf, t);
}

// ---------------- LayerNorm 1 + context-token finalization ----------------
__global__ __launch_bounds__(32)
void ln1_kernel(const float* __restrict__ x,
                const float* __restrict__ ln_w, const float* __restrict__ ln_b,
                const float* __restrict__ gamma1,
                float* __restrict__ out,
                unsigned short* __restrict__ xn_bf,
                unsigned short* __restrict__ xq_bf) {
  const int t = blockIdx.x;            // 0 .. BN-1
  const int lane = threadIdx.x;
  const float* xr = x + (size_t)t * Cc;
  float v[8];
  float s = 0.f;
  #pragma unroll
  for (int i = 0; i < 8; ++i) { v[i] = xr[lane * 8 + i]; s += v[i]; }
  #pragma unroll
  for (int off = 16; off; off >>= 1) s += __shfl_xor(s, off, 32);
  const float mean = s * (1.0f / Cc);
  float ss = 0.f;
  #pragma unroll
  for (int i = 0; i < 8; ++i) { float d = v[i] - mean; ss += d * d; }
  #pragma unroll
  for (int off = 16; off; off >>= 1) ss += __shfl_xor(ss, off, 32);
  const float rs = rsqrtf(ss * (1.0f / Cc) + 1e-5f);

  const int b = t / NTOK, n = t % NTOK;
  #pragma unroll
  for (int i = 0; i < 8; ++i) {
    const int c = lane * 8 + i;
    const float xn = (v[i] - mean) * rs * ln_w[c] + ln_b[c];
    xn_bf[(size_t)t * Cc + c] = f2bf(xn);
    if (n < Qq) {
      xq_bf[((size_t)(b * Qq + n)) * Cc + c] = f2bf(xn);
    } else {
      // context tokens: reference double-adds them -> out = 2*(x + g1*xn)
      out[(size_t)t * Cc + c] = 2.0f * (v[i] + gamma1[c] * xn);
    }
  }
}

// ---------------- LayerNorm 2 (query rows only) ----------------
__global__ __launch_bounds__(32)
void ln2_kernel(const float* __restrict__ x1,
                const float* __restrict__ ln_w, const float* __restrict__ ln_b,
                float* __restrict__ y, unsigned short* __restrict__ y_bf) {
  const int t = blockIdx.x;            // 0 .. BQ-1
  const int lane = threadIdx.x;
  const float* xr = x1 + (size_t)t * Cc;
  float v[8];
  float s = 0.f;
  #pragma unroll
  for (int i = 0; i < 8; ++i) { v[i] = xr[lane * 8 + i]; s += v[i]; }
  #pragma unroll
  for (int off = 16; off; off >>= 1) s += __shfl_xor(s, off, 32);
  const float mean = s * (1.0f / Cc);
  float ss = 0.f;
  #pragma unroll
  for (int i = 0; i < 8; ++i) { float d = v[i] - mean; ss += d * d; }
  #pragma unroll
  for (int off = 16; off; off >>= 1) ss += __shfl_xor(ss, off, 32);
  const float rs = rsqrtf(ss * (1.0f / Cc) + 1e-5f);
  #pragma unroll
  for (int i = 0; i < 8; ++i) {
    const int c = lane * 8 + i;
    const float yn = (v[i] - mean) * rs * ln_w[c] + ln_b[c];
    y[(size_t)t * Cc + c] = yn;
    y_bf[(size_t)t * Cc + c] = f2bf(yn);
  }
}

// ---------------- weight fp32 [in][out] -> bf16 transposed [out][in] ----------------
__global__ void wT_kernel(const float* __restrict__ src, unsigned short* __restrict__ dst,
                          int R, int Cn) {
  const int i = blockIdx.x * blockDim.x + threadIdx.x;
  if (i >= R * Cn) return;
  const int r = i / Cn, c = i % Cn;
  dst[(size_t)c * R + r] = f2bf(src[i]);
}

// ---------------- bf16 WMMA GEMM, 2x2 register blocking + async LDS A-panel -------
// Block: 128 threads (4 waves). Block tile: M=32 x N=128.
// Wave w computes a 32x32 output tile (2x2 of 16x16 WMMA tiles) at columns
// n0w = blockIdx.y*128 + w*32. All 4 waves share the same 32 M-rows; the 32x32
// A k-panel is staged per k-step with GLOBAL_LOAD_ASYNC_TO_LDS_B128 (memory->LDS
// direct, ASYNCcnt-tracked, no VGPR bounce); each wave moves 32 lanes x 16B.
#define MODE_Q    1
#define MODE_K    2
#define MODE_VT   3
#define MODE_GELU 4
#define MODE_EPI  5

__global__ __launch_bounds__(128)
void gemm_wmma_bf16(const unsigned short* __restrict__ A,
                    const unsigned short* __restrict__ BT,
                    int M, int Nn, int K, int mode, float scale,
                    const float* __restrict__ res,
                    const float* __restrict__ gamma,
                    int res_remap, int out_remap,
                    float* __restrict__ outf,
                    unsigned short* __restrict__ outbf) {
  (void)M;
  __shared__ __attribute__((aligned(16))) unsigned short Alds[32 * ALDS_STRIDE];

  const int tid  = threadIdx.x;
  const int lane = tid & 31;
  const int w    = tid >> 5;
  const int m0   = blockIdx.x * 32;
  const int n0w  = blockIdx.y * 128 + w * 32;

  // A staging assignment: thread -> (row, 8-element chunk); 128 threads x 16B = panel
  const int arow = tid >> 2;            // 0..31
  const int achk = (tid & 3) * 8;       // 0,8,16,24
  const unsigned short* pa = A + (size_t)(m0 + arow) * K + achk;
  // generic pointers to __shared__ carry the LDS byte offset in their low 32 bits
  const unsigned int ldsoff =
      (unsigned int)(unsigned long long)(&Alds[arow * ALDS_STRIDE + achk]);

  const unsigned short* pb = BT + (size_t)n0w * K;

  v8f acc[2][2] = {{{}, {}}, {{}, {}}};

  for (int k0 = 0; k0 < K; k0 += 32) {
    __builtin_prefetch(pb + k0 + 128, 0, 1);
    __syncthreads();                       // previous iteration's frag reads done
    // async DMA: 16B per lane, memory -> LDS, tracked by ASYNCcnt
    asm volatile("global_load_async_to_lds_b128 %0, %1, off"
                 :: "v"(ldsoff), "v"(pa + k0) : "memory");
    asm volatile("s_wait_asynccnt 0x0" ::: "memory");   // my slice landed
    __syncthreads();                       // everyone's slice landed

    v16bf a0 = load_frag(Alds, ALDS_STRIDE, lane);
    v16bf a1 = load_frag(Alds + 16 * ALDS_STRIDE, ALDS_STRIDE, lane);
    v16bf b0 = load_frag(pb + k0, K, lane);
    v16bf b1 = load_frag(pb + (size_t)16 * K + k0, K, lane);

    acc[0][0] = __builtin_amdgcn_wmma_f32_16x16x32_bf16(false, a0, false, b0, (short)0, acc[0][0], false, false);
    acc[0][1] = __builtin_amdgcn_wmma_f32_16x16x32_bf16(false, a0, false, b1, (short)0, acc[0][1], false, false);
    acc[1][0] = __builtin_amdgcn_wmma_f32_16x16x32_bf16(false, a1, false, b0, (short)0, acc[1][0], false, false);
    acc[1][1] = __builtin_amdgcn_wmma_f32_16x16x32_bf16(false, a1, false, b1, (short)0, acc[1][1], false, false);
  }

  // C layout per 16x16 tile: VGPR r: lanes 0-15 -> row +r, lanes 16-31 -> row +8+r;
  // col = +(lane&15)
  const int keycol = lane & 15;
  const int rowhf  = (lane >> 4) * 8;

  if (mode == MODE_Q) {
    #pragma unroll
    for (int i = 0; i < 2; ++i)
    #pragma unroll
    for (int j = 0; j < 2; ++j) {
      const int colc = n0w + j * 16 + keycol;
      const int hh = colc >> 5, dd = colc & 31;
      #pragma unroll
      for (int r = 0; r < 8; ++r) {
        const int row = m0 + i * 16 + rowhf + r;
        const int bidx = row / Qq, qi = row % Qq;
        const size_t idx = (((size_t)(bidx * Hh + hh) * QPAD) + qi) * Dd + dd;
        outbf[idx] = f2bf(acc[i][j][r] * scale);
      }
    }
  } else if (mode == MODE_K) {
    #pragma unroll
    for (int i = 0; i < 2; ++i)
    #pragma unroll
    for (int j = 0; j < 2; ++j) {
      const int colc = n0w + j * 16 + keycol;
      const int hh = colc >> 5, dd = colc & 31;
      #pragma unroll
      for (int r = 0; r < 8; ++r) {
        const int row = m0 + i * 16 + rowhf + r;
        const int bidx = row / NTOK, n = row % NTOK;
        const size_t idx = (((size_t)(bidx * Hh + hh) * NPAD) + n) * Dd + dd;
        outbf[idx] = f2bf(acc[i][j][r]);
      }
    }
  } else if (mode == MODE_VT) {
    #pragma unroll
    for (int i = 0; i < 2; ++i)
    #pragma unroll
    for (int j = 0; j < 2; ++j) {
      const int colc = n0w + j * 16 + keycol;
      const int hh = colc >> 5, dd = colc & 31;
      #pragma unroll
      for (int r = 0; r < 8; ++r) {
        const int row = m0 + i * 16 + rowhf + r;
        const int bidx = row / NTOK, n = row % NTOK;
        const size_t idx = (((size_t)(bidx * Hh + hh) * Dd) + dd) * NPAD + n;
        outbf[idx] = f2bf(acc[i][j][r]);
      }
    }
  } else if (mode == MODE_GELU) {
    #pragma unroll
    for (int i = 0; i < 2; ++i)
    #pragma unroll
    for (int j = 0; j < 2; ++j) {
      const int colc = n0w + j * 16 + keycol;
      #pragma unroll
      for (int r = 0; r < 8; ++r) {
        const int row = m0 + i * 16 + rowhf + r;
        const float v = acc[i][j][r];
        const float g = 0.5f * v * (1.0f + erff(v * 0.70710678118654752f));
        outbf[(size_t)row * Nn + colc] = f2bf(g);
      }
    }
  } else { // MODE_EPI : out = res + gamma*acc, optional B*Q->B*N row remap
    #pragma unroll
    for (int i = 0; i < 2; ++i)
    #pragma unroll
    for (int j = 0; j < 2; ++j) {
      const int colc = n0w + j * 16 + keycol;
      #pragma unroll
      for (int r = 0; r < 8; ++r) {
        const int row = m0 + i * 16 + rowhf + r;
        const size_t rr = res_remap ? ((size_t)(row / Qq) * NTOK + (row % Qq)) : (size_t)row;
        const size_t ro = out_remap ? ((size_t)(row / Qq) * NTOK + (row % Qq)) : (size_t)row;
        outf[ro * Nn + colc] = res[rr * Nn + colc] + gamma[colc] * acc[i][j][r];
      }
    }
  }
}

// ---------------- flash attention: one wave per (b, h, 16-query tile) ----------------
__global__ __launch_bounds__(32)
void attn_kernel(const unsigned short* __restrict__ q_bf,   // [B,H,QPAD,32]
                 const unsigned short* __restrict__ k_bf,   // [B,H,NPAD,32]
                 const unsigned short* __restrict__ vT_bf,  // [B,H,32,NPAD]
                 unsigned short* __restrict__ O_bf) {       // [B*Q, C] row-major
  __shared__ __attribute__((aligned(16))) unsigned short Plds[16 * 32];
  const int lane = threadIdx.x;
  const int qt = blockIdx.x, h = blockIdx.y, b = blockIdx.z;
  const unsigned short* qp = q_bf  + ((size_t)(b * Hh + h) * QPAD + qt * 16) * Dd;
  const unsigned short* kp = k_bf  + (size_t)(b * Hh + h) * NPAD * Dd;
  const unsigned short* vp = vT_bf + (size_t)(b * Hh + h) * Dd * NPAD;

  const v16bf qf = load_frag(qp, Dd, lane);
  const int keycol = lane & 15;
  const int hf = lane >> 4;

  float mrow[8], lrow[8];
  v8f o0 = {}, o1 = {};
  #pragma unroll
  for (int r = 0; r < 8; ++r) { mrow[r] = -3.0e38f; lrow[r] = 0.f; }

  for (int n0 = 0; n0 < NPAD; n0 += 32) {
    v16bf kf0 = load_frag(kp + (size_t)n0 * Dd, Dd, lane);
    v16bf kf1 = load_frag(kp + (size_t)(n0 + 16) * Dd, Dd, lane);
    v8f z = {};
    v8f s0 = __builtin_amdgcn_wmma_f32_16x16x32_bf16(false, qf, false, kf0, (short)0, z, false, false);
    v8f s1 = __builtin_amdgcn_wmma_f32_16x16x32_bf16(false, qf, false, kf1, (short)0, z, false, false);

    const bool ok0 = (n0 + keycol) < NTOK;
    const bool ok1 = (n0 + 16 + keycol) < NTOK;

    #pragma unroll
    for (int r = 0; r < 8; ++r) {
      float a  = ok0 ? s0[r] : -1.0e30f;
      float c2 = ok1 ? s1[r] : -1.0e30f;
      float mx = fmaxf(a, c2);
      mx = fmaxf(mx, __shfl_xor(mx, 1, 16));
      mx = fmaxf(mx, __shfl_xor(mx, 2, 16));
      mx = fmaxf(mx, __shfl_xor(mx, 4, 16));
      mx = fmaxf(mx, __shfl_xor(mx, 8, 16));
      const float mnew = fmaxf(mrow[r], mx);
      const float corr = __expf(mrow[r] - mnew);
      const float p0 = __expf(a - mnew);
      const float p1 = __expf(c2 - mnew);
      float ps = p0 + p1;
      ps += __shfl_xor(ps, 1, 16);
      ps += __shfl_xor(ps, 2, 16);
      ps += __shfl_xor(ps, 4, 16);
      ps += __shfl_xor(ps, 8, 16);
      lrow[r] = lrow[r] * corr + ps;
      o0[r] *= corr;
      o1[r] *= corr;
      mrow[r] = mnew;
      // C-layout -> LDS (row-major P tile, 16x32 bf16) for A-fragment reload
      Plds[(hf * 8 + r) * 32 + keycol]      = f2bf(p0);
      Plds[(hf * 8 + r) * 32 + 16 + keycol] = f2bf(p1);
    }
    __syncthreads();

    v16bf pf  = load_frag(Plds, 32, lane);
    v16bf vf0 = load_frag(vp + n0, NPAD, lane);
    v16bf vf1 = load_frag(vp + (size_t)16 * NPAD + n0, NPAD, lane);
    o0 = __builtin_amdgcn_wmma_f32_16x16x32_bf16(false, pf, false, vf0, (short)0, o0, false, false);
    o1 = __builtin_amdgcn_wmma_f32_16x16x32_bf16(false, pf, false, vf1, (short)0, o1, false, false);
    __syncthreads();
  }

  #pragma unroll
  for (int r = 0; r < 8; ++r) {
    const int qg = qt * 16 + hf * 8 + r;
    if (qg < Qq) {
      const float inv = 1.0f / lrow[r];
      const size_t o = ((size_t)(b * Qq + qg)) * Cc + h * Dd;
      O_bf[o + keycol]      = f2bf(o0[r] * inv);
      O_bf[o + 16 + keycol] = f2bf(o1[r] * inv);
    }
  }
}

// ---------------- host orchestration ----------------
extern "C" void kernel_launch(void* const* d_in, const int* in_sizes, int n_in,
                              void* d_out, int out_size, void* d_ws, size_t ws_size,
                              hipStream_t stream) {
  (void)in_sizes; (void)n_in; (void)out_size; (void)ws_size;
  const float* x      = (const float*)d_in[0];
  const float* Wq     = (const float*)d_in[1];
  const float* Wk     = (const float*)d_in[2];
  const float* Wv     = (const float*)d_in[3];
  const float* Wp     = (const float*)d_in[4];
  const float* W1     = (const float*)d_in[5];
  const float* W2     = (const float*)d_in[6];
  const float* ln1_w  = (const float*)d_in[7];
  const float* ln1_b  = (const float*)d_in[8];
  const float* ln2_w  = (const float*)d_in[9];
  const float* ln2_b  = (const float*)d_in[10];
  const float* gamma1 = (const float*)d_in[11];
  const float* gamma2 = (const float*)d_in[12];
  float* out = (float*)d_out;

  char* ws = (char*)d_ws;
  size_t off = 0;
  auto alloc = [&](size_t bytes) -> void* {
    void* p = ws + off;
    off += (bytes + 255) & ~(size_t)255;
    return p;
  };

  unsigned short* xn_bf = (unsigned short*)alloc((size_t)BN * Cc * 2);
  unsigned short* xq_bf = (unsigned short*)alloc((size_t)BQ * Cc * 2);
  unsigned short* WqT   = (unsigned short*)alloc((size_t)Cc * Cc * 2);
  unsigned short* WkT   = (unsigned short*)alloc((size_t)Cc * Cc * 2);
  unsigned short* WvT   = (unsigned short*)alloc((size_t)Cc * Cc * 2);
  unsigned short* WpT   = (unsigned short*)alloc((size_t)Cc * Cc * 2);
  unsigned short* W1T   = (unsigned short*)alloc((size_t)Cc * HIDN * 2);
  unsigned short* W2T   = (unsigned short*)alloc((size_t)HIDN * Cc * 2);
  unsigned short* q_bf  = (unsigned short*)alloc((size_t)Bb * Hh * QPAD * Dd * 2);
  unsigned short* k_bf  = (unsigned short*)alloc((size_t)Bb * Hh * NPAD * Dd * 2);
  unsigned short* vT_bf = (unsigned short*)alloc((size_t)Bb * Hh * NPAD * Dd * 2);
  unsigned short* O_bf  = (unsigned short*)alloc((size_t)BQ * Cc * 2);
  float*          x1    = (float*)alloc((size_t)BQ * Cc * 4);
  float*          yv    = (float*)alloc((size_t)BQ * Cc * 4);
  unsigned short* y_bf  = (unsigned short*)alloc((size_t)BQ * Cc * 2);
  unsigned short* h_bf  = (unsigned short*)alloc((size_t)BQ * HIDN * 2);

  // 1) weights -> bf16 transposed [out][in]
  wT_kernel<<<(Cc * Cc + 255) / 256, 256, 0, stream>>>(Wq, WqT, Cc, Cc);
  wT_kernel<<<(Cc * Cc + 255) / 256, 256, 0, stream>>>(Wk, WkT, Cc, Cc);
  wT_kernel<<<(Cc * Cc + 255) / 256, 256, 0, stream>>>(Wv, WvT, Cc, Cc);
  wT_kernel<<<(Cc * Cc + 255) / 256, 256, 0, stream>>>(Wp, WpT, Cc, Cc);
  wT_kernel<<<(Cc * HIDN + 255) / 256, 256, 0, stream>>>(W1, W1T, Cc, HIDN);
  wT_kernel<<<(HIDN * Cc + 255) / 256, 256, 0, stream>>>(W2, W2T, HIDN, Cc);

  // 2) LN1 (+ finalize context rows of d_out)
  ln1_kernel<<<BN, 32, 0, stream>>>(x, ln1_w, ln1_b, gamma1, out, xn_bf, xq_bf);

  const float scaleq = 0.17677669529663687f; // D^-0.5

  // 3) projections (block tile M=32 x N=128)
  gemm_wmma_bf16<<<dim3(BQ / 32, Cc / 128), 128, 0, stream>>>(
      xq_bf, WqT, BQ, Cc, Cc, MODE_Q, scaleq, nullptr, nullptr, 0, 0, nullptr, q_bf);
  gemm_wmma_bf16<<<dim3(BN / 32, Cc / 128), 128, 0, stream>>>(
      xn_bf, WkT, BN, Cc, Cc, MODE_K, 1.0f, nullptr, nullptr, 0, 0, nullptr, k_bf);
  gemm_wmma_bf16<<<dim3(BN / 32, Cc / 128), 128, 0, stream>>>(
      xn_bf, WvT, BN, Cc, Cc, MODE_VT, 1.0f, nullptr, nullptr, 0, 0, nullptr, vT_bf);

  // 4) flash attention
  attn_kernel<<<dim3(QTILES, Hh, Bb), 32, 0, stream>>>(q_bf, k_bf, vT_bf, O_bf);

  // 5) output projection + residual (x1 = x[:, :Q] + g1 * (O @ Wp))
  gemm_wmma_bf16<<<dim3(BQ / 32, Cc / 128), 128, 0, stream>>>(
      O_bf, WpT, BQ, Cc, Cc, MODE_EPI, 1.0f, x, gamma1, 1, 0, x1, nullptr);

  // 6) LN2 on query rows
  ln2_kernel<<<BQ, 32, 0, stream>>>(x1, ln2_w, ln2_b, yv, y_bf);

  // 7) MLP: h = gelu(y @ W1) ; out_q = y + g2 * (h @ W2)
  gemm_wmma_bf16<<<dim3(BQ / 32, HIDN / 128), 128, 0, stream>>>(
      y_bf, W1T, BQ, HIDN, Cc, MODE_GELU, 1.0f, nullptr, nullptr, 0, 0, nullptr, h_bf);
  gemm_wmma_bf16<<<dim3(BQ / 32, Cc / 128), 128, 0, stream>>>(
      h_bf, W2T, BQ, Cc, HIDN, MODE_EPI, 1.0f, yv, gamma2, 0, 1, out, nullptr);
}
